// CapsuleNet_65146063946634
// MI455X (gfx1250) — compile-verified, matrix operand order checked by reading, not run on hardware
//
#include <hip/hip_runtime.h>
#include <hip/hip_bf16.h>
#include <math.h>

// ---------------------------------------------------------------------------
// CapsuleNet forward for gfx1250 (MI455X), bf16 WMMA path with f32 accumulate.
// ---------------------------------------------------------------------------

typedef __attribute__((ext_vector_type(16))) __bf16 v16bf;
typedef __attribute__((ext_vector_type(8)))  float  v8f;

static __device__ __forceinline__ v8f wmma_bf16(v16bf a, v16bf b, v8f c) {
    // (neg_a, A, neg_b, B, c_mod, C, reuse_a, reuse_b)
    return __builtin_amdgcn_wmma_f32_16x16x32_bf16(false, a, false, b, (short)0, c,
                                                   false, false);
}

// ---------------------------------------------------------------------------
// Workspace layout (bytes)
// ---------------------------------------------------------------------------
#define OFF_Y   ((size_t)0)            // y bf16: 512*256*400*2      = 104857600
#define OFF_W2  ((size_t)104857600)    // conv2 W frags bf16         =  12582912
#define OFF_W1  ((size_t)117440512)    // conv1 W frags bf16         =     49152
#define OFF_U   ((size_t)117489664)    // u f32: 512*1152*8*4        =  18874368
#define OFF_LG  ((size_t)136364032)    // logits f32: 512*10*4       =     20480

// ---------------------------------------------------------------------------
// Kernel 1: pack conv1 weights (256,1,9,9) f32 -> B fragments, K padded 81->96
// fragment group index = c*16 + ntile ; per group 32 lanes x 16 bf16
// ---------------------------------------------------------------------------
__global__ void prep_w1(const float* __restrict__ conv_w, __bf16* __restrict__ w1f) {
    int gid = blockIdx.x * blockDim.x + threadIdx.x;       // 0 .. 3*16*32-1
    if (gid >= 3 * 16 * 32) return;
    int lane  = gid & 31;
    int ntile = (gid >> 5) & 15;
    int c     = gid >> 9;                                  // 0..2
    int col   = lane & 15;
    int half  = lane >> 4;
    int n     = ntile * 16 + col;
    __bf16* dst = w1f + (size_t)gid * 16;
#pragma unroll
    for (int j = 0; j < 8; ++j) {
        int k0 = 32 * c + 8 * half + j;
        int k1 = k0 + 16;
        dst[j]     = (k0 < 81) ? (__bf16)conv_w[n * 81 + k0] : (__bf16)0.0f;
        dst[8 + j] = (k1 < 81) ? (__bf16)conv_w[n * 81 + k1] : (__bf16)0.0f;
    }
}

// ---------------------------------------------------------------------------
// Kernel 2: pack conv2 weights (256,256,9,9) f32 -> B fragments.
// group index = (cin*3 + c)*16 + ntile
// ---------------------------------------------------------------------------
__global__ void prep_w2(const float* __restrict__ prim_w, __bf16* __restrict__ w2f) {
    int gid = blockIdx.x * blockDim.x + threadIdx.x;       // 0 .. 256*3*16*32-1
    if (gid >= 256 * 3 * 16 * 32) return;
    int lane  = gid & 31;
    int ntile = (gid >> 5) & 15;
    int g     = gid >> 9;                                  // cin*3 + c
    int c     = g % 3;
    int cin   = g / 3;
    int col   = lane & 15;
    int half  = lane >> 4;
    int n     = ntile * 16 + col;
    const float* src = prim_w + ((size_t)n * 256 + cin) * 81;
    __bf16* dst = w2f + (size_t)gid * 16;
#pragma unroll
    for (int j = 0; j < 8; ++j) {
        int k0 = 32 * c + 8 * half + j;
        int k1 = k0 + 16;
        dst[j]     = (k0 < 81) ? (__bf16)src[k0] : (__bf16)0.0f;
        dst[8 + j] = (k1 < 81) ? (__bf16)src[k1] : (__bf16)0.0f;
    }
}

// ---------------------------------------------------------------------------
// Kernel 3: conv1 as implicit GEMM.  M = 512*20*20 = 204800, N = 256, K = 96.
// One wave computes 16 rows x 64 cols (4 n-tiles), 12 WMMAs.
// Output y stored bf16, NCHW.
// ---------------------------------------------------------------------------
__global__ __launch_bounds__(256) void conv1_wmma(const float* __restrict__ x,
                                                  const float* __restrict__ conv_b,
                                                  const __bf16* __restrict__ w1f,
                                                  __bf16* __restrict__ y) {
    int wave = threadIdx.x >> 5;
    int lane = threadIdx.x & 31;
    int gid  = blockIdx.x * 8 + wave;          // 0 .. 51199
    int mtile = gid >> 2;                      // 0 .. 12799
    int ngrp  = gid & 3;
    int col  = lane & 15;
    int half = lane >> 4;

    // A fragment rows: m = mtile*16 + (lane&15)
    int m   = mtile * 16 + col;
    int b   = m / 400;
    int rem = m - b * 400;
    int oy  = rem / 20;
    int ox  = rem - oy * 20;
    const float* xb = x + (size_t)b * 784;

    v16bf a[3];
#pragma unroll
    for (int c = 0; c < 3; ++c) {
#pragma unroll
        for (int j = 0; j < 8; ++j) {
            int k0 = 32 * c + 8 * half + j;
            int k1 = k0 + 16;
            float f0 = 0.0f, f1 = 0.0f;
            if (k0 < 81) f0 = xb[(oy + k0 / 9) * 28 + ox + k0 % 9];
            if (k1 < 81) f1 = xb[(oy + k1 / 9) * 28 + ox + k1 % 9];
            a[c][j]     = (__bf16)f0;
            a[c][8 + j] = (__bf16)f1;
        }
    }

#pragma unroll
    for (int t = 0; t < 4; ++t) {
        int ntile = ngrp * 4 + t;
        v8f acc = {};
#pragma unroll
        for (int c = 0; c < 3; ++c) {
            v16bf bf = *(const v16bf*)(w1f + ((size_t)(c * 16 + ntile) * 32 + lane) * 16);
            acc = wmma_bf16(a[c], bf, acc);
        }
        int n = ntile * 16 + col;              // C/D: N = lane&15
        float bias = conv_b[n];
#pragma unroll
        for (int v = 0; v < 8; ++v) {
            int mr = mtile * 16 + v + 8 * half;  // C/D: M = vgpr + 8*half
            int bb = mr / 400;
            int rr = mr - bb * 400;
            y[((size_t)bb * 256 + n) * 400 + rr] = (__bf16)(acc[v] + bias);
        }
    }
}

// ---------------------------------------------------------------------------
// Kernel 4: conv2 (primary caps) implicit GEMM + fused squash -> u.
// One block per image b.  M = 36 (pad 48 -> 3 m-tiles), N = 256, K = 256*96.
// 8 waves: wave w owns n-tiles {2w, 2w+1}; 32 input channels staged in LDS
// per outer step.  Epilogue writes p tile to LDS, computes u = squash.
// ---------------------------------------------------------------------------
__global__ __launch_bounds__(256) void conv2_wmma(const __bf16* __restrict__ y,
                                                  const float* __restrict__ prim_b,
                                                  const __bf16* __restrict__ w2f,
                                                  float* __restrict__ u) {
    __shared__ __align__(16) unsigned char smem[36864];    // union: stage / p
    __bf16* ylds = (__bf16*)smem;                          // 32 * 400 bf16
    float*  plds = (float*)smem;                           // 36 * 256 f32

    int b    = blockIdx.x;
    int tid  = threadIdx.x;
    int wave = tid >> 5;
    int lane = tid & 31;
    int col  = lane & 15;
    int half = lane >> 4;

    v8f zero = {};
    v8f acc[3][2];
#pragma unroll
    for (int mt = 0; mt < 3; ++mt)
#pragma unroll
        for (int nt = 0; nt < 2; ++nt) acc[mt][nt] = zero;

    for (int cb = 0; cb < 8; ++cb) {
        __syncthreads();
        // stage y[b, cb*32 .. cb*32+31, :, :] -> LDS (12800 bf16 = 6400 u32)
        const unsigned int* src =
            (const unsigned int*)(y + ((size_t)b * 256 + cb * 32) * 400);
        unsigned int* dst = (unsigned int*)smem;
        for (int i = tid; i < 6400; i += 256) dst[i] = src[i];
        __syncthreads();

        for (int cl = 0; cl < 32; ++cl) {
            int cin = cb * 32 + cl;
            const __bf16* yl = ylds + cl * 400;
#pragma unroll
            for (int c = 0; c < 3; ++c) {
                v16bf bf0 = *(const v16bf*)(w2f +
                    ((size_t)(((cin * 3 + c) * 16 + wave * 2 + 0) * 32 + lane)) * 16);
                v16bf bf1 = *(const v16bf*)(w2f +
                    ((size_t)(((cin * 3 + c) * 16 + wave * 2 + 1) * 32 + lane)) * 16);
#pragma unroll
                for (int mt = 0; mt < 3; ++mt) {
                    int m = mt * 16 + col;               // output position row
                    bool valid = (m < 36);
                    int oy = valid ? (m / 6) : 0;
                    int ox = valid ? (m - oy * 6) : 0;
                    v16bf af;
#pragma unroll
                    for (int j = 0; j < 8; ++j) {
                        int k0 = 32 * c + 8 * half + j;
                        int k1 = k0 + 16;
                        float f0 = 0.0f, f1 = 0.0f;
                        if (valid && k0 < 81)
                            f0 = (float)yl[(2 * oy + k0 / 9) * 20 + 2 * ox + k0 % 9];
                        if (valid && k1 < 81)
                            f1 = (float)yl[(2 * oy + k1 / 9) * 20 + 2 * ox + k1 % 9];
                        af[j]     = (__bf16)f0;
                        af[8 + j] = (__bf16)f1;
                    }
                    acc[mt][0] = wmma_bf16(af, bf0, acc[mt][0]);
                    acc[mt][1] = wmma_bf16(af, bf1, acc[mt][1]);
                }
            }
        }
    }
    __syncthreads();

    // p tile -> LDS (with bias)
#pragma unroll
    for (int mt = 0; mt < 3; ++mt) {
#pragma unroll
        for (int nt = 0; nt < 2; ++nt) {
            int n = (wave * 2 + nt) * 16 + col;
            float bias = prim_b[n];
#pragma unroll
            for (int v = 0; v < 8; ++v) {
                int m = mt * 16 + v + 8 * half;
                if (m < 36) plds[m * 256 + n] = acc[mt][nt][v] + bias;
            }
        }
    }
    __syncthreads();

    // u[b, r, i] = squash over i of p[b, i*32 + r32, s], r = r32*36 + s
    for (int r = tid; r < 1152; r += 256) {
        int r32 = r / 36;
        int m   = r - r32 * 36;
        float t[8], sn = 0.0f;
#pragma unroll
        for (int i = 0; i < 8; ++i) {
            t[i] = plds[m * 256 + i * 32 + r32];
            sn += t[i] * t[i];
        }
        float scale = sn / ((1.0f + sn) * sqrtf(sn));
        float* ur = u + ((size_t)b * 1152 + r) * 8;
#pragma unroll
        for (int i = 0; i < 8; ++i) ur[i] = t[i] * scale;
    }
}

// ---------------------------------------------------------------------------
// Kernel 5: dynamic routing.  One block per (b, class c).
// priors (1152x16) kept bf16 in LDS; 3 iterations fully in LDS.
// ---------------------------------------------------------------------------
__global__ __launch_bounds__(256) void routing(const float* __restrict__ u,
                                               const float* __restrict__ route_w,
                                               float* __restrict__ logits) {
    __shared__ __bf16 pri[1152 * 16];       // 36864 B
    __shared__ float  blog[1152];
    __shared__ float  probs[1152];
    __shared__ float  red[256];
    __shared__ float  sv[16];

    int bc = blockIdx.x;
    int b  = bc / 10;
    int c  = bc - b * 10;
    int tid = threadIdx.x;

    // priors[r, o] = sum_i u[b,r,i] * route_w[c,r,i,o]
    for (int r = tid; r < 1152; r += 256) {
        float ur[8];
        const float* up = u + ((size_t)b * 1152 + r) * 8;
#pragma unroll
        for (int i = 0; i < 8; ++i) ur[i] = up[i];
        const float* w = route_w + ((size_t)(c * 1152 + r)) * 128;
#pragma unroll
        for (int o = 0; o < 16; ++o) {
            float s = 0.0f;
#pragma unroll
            for (int i = 0; i < 8; ++i) s += ur[i] * w[i * 16 + o];
            pri[r * 16 + o] = (__bf16)s;
        }
        blog[r] = 0.0f;
    }
    __syncthreads();

    for (int it = 0; it < 3; ++it) {
        // softmax over r of blog
        float mx = -INFINITY;
        for (int r = tid; r < 1152; r += 256) mx = fmaxf(mx, blog[r]);
        red[tid] = mx;
        __syncthreads();
        for (int s = 128; s > 0; s >>= 1) {
            if (tid < s) red[tid] = fmaxf(red[tid], red[tid + s]);
            __syncthreads();
        }
        mx = red[0];
        __syncthreads();

        float sum = 0.0f;
        for (int r = tid; r < 1152; r += 256) {
            float e = __expf(blog[r] - mx);
            probs[r] = e;
            sum += e;
        }
        red[tid] = sum;
        __syncthreads();
        for (int s = 128; s > 0; s >>= 1) {
            if (tid < s) red[tid] += red[tid + s];
            __syncthreads();
        }
        float inv = 1.0f / red[0];
        __syncthreads();

        // s[o] = inv * sum_r pri[r,o] * probs[r]
        int o  = tid >> 4;
        int rs = tid & 15;
        float part = 0.0f;
        for (int r = rs; r < 1152; r += 16)
            part += (float)pri[r * 16 + o] * probs[r];
        red[o * 16 + rs] = part;
        __syncthreads();
        if (tid < 16) {
            float s = 0.0f;
#pragma unroll
            for (int k = 0; k < 16; ++k) s += red[tid * 16 + k];
            sv[tid] = s * inv;
        }
        __syncthreads();

        // v = squash(s)   (wave-0 lockstep: reads before writes)
        if (tid < 16) {
            float sn = 0.0f;
#pragma unroll
            for (int o2 = 0; o2 < 16; ++o2) sn += sv[o2] * sv[o2];
            float scale = sn / ((1.0f + sn) * sqrtf(sn));
            sv[tid] = sv[tid] * scale;
        }
        __syncthreads();

        // blog[r] += sum_o pri[r,o] * v[o]
        for (int r = tid; r < 1152; r += 256) {
            float d = 0.0f;
#pragma unroll
            for (int o2 = 0; o2 < 16; ++o2) d += (float)pri[r * 16 + o2] * sv[o2];
            blog[r] += d;
        }
        __syncthreads();
    }

    if (tid == 0) {
        float sn = 0.0f;
#pragma unroll
        for (int o2 = 0; o2 < 16; ++o2) sn += sv[o2] * sv[o2];
        logits[bc] = sqrtf(sn);
    }
}

// ---------------------------------------------------------------------------
// Kernel 6: softmax over 10 classes -> d_out (512,10) f32
// ---------------------------------------------------------------------------
__global__ void final_softmax(const float* __restrict__ logits, float* __restrict__ out) {
    int b = blockIdx.x * blockDim.x + threadIdx.x;
    if (b >= 512) return;
    float l[10];
    float mx = -INFINITY;
#pragma unroll
    for (int c = 0; c < 10; ++c) {
        l[c] = logits[b * 10 + c];
        mx = fmaxf(mx, l[c]);
    }
    float s = 0.0f;
#pragma unroll
    for (int c = 0; c < 10; ++c) {
        l[c] = __expf(l[c] - mx);
        s += l[c];
    }
#pragma unroll
    for (int c = 0; c < 10; ++c) out[b * 10 + c] = l[c] / s;
}

// ---------------------------------------------------------------------------
// Host launcher
// ---------------------------------------------------------------------------
extern "C" void kernel_launch(void* const* d_in, const int* in_sizes, int n_in,
                              void* d_out, int out_size, void* d_ws, size_t ws_size,
                              hipStream_t stream) {
    const float* x       = (const float*)d_in[0];   // (512,1,28,28)
    const float* conv_w  = (const float*)d_in[1];   // (256,1,9,9)
    const float* conv_b  = (const float*)d_in[2];   // (256,)
    const float* prim_w  = (const float*)d_in[3];   // (256,256,9,9)
    const float* prim_b  = (const float*)d_in[4];   // (256,)
    const float* route_w = (const float*)d_in[5];   // (10,1152,8,16)
    float* out = (float*)d_out;                     // (512,10)

    char* ws = (char*)d_ws;
    __bf16* y_ws   = (__bf16*)(ws + OFF_Y);
    __bf16* w2f_ws = (__bf16*)(ws + OFF_W2);
    __bf16* w1f_ws = (__bf16*)(ws + OFF_W1);
    float*  u_ws   = (float*)(ws + OFF_U);
    float*  lg_ws  = (float*)(ws + OFF_LG);

    prep_w1<<<6, 256, 0, stream>>>(conv_w, w1f_ws);
    prep_w2<<<1536, 256, 0, stream>>>(prim_w, w2f_ws);
    conv1_wmma<<<6400, 256, 0, stream>>>(x, conv_b, w1f_ws, y_ws);
    conv2_wmma<<<512, 256, 0, stream>>>(y_ws, prim_b, w2f_ws, u_ws);
    routing<<<5120, 256, 0, stream>>>(u_ws, route_w, lg_ws);
    final_softmax<<<2, 256, 0, stream>>>(lg_ws, out);
}
